// KANC_HDC_42159398977551
// MI455X (gfx1250) — 8-bit matrix operand order repaired and checked by reading, compiled, not run on hardware
//
#include <hip/hip_runtime.h>
#include <stdint.h>

typedef __attribute__((ext_vector_type(8))) int v8i;

#define BN    4096
#define FN    200
#define DN    10000
#define CN    10
#define BINS  16
#define NWP   320
#define NCHK  157
#define BND_STRIDE 322

__global__ void hdc_pack_kernel(const float* __restrict__ cbf,
                                const float* __restrict__ chf,
                                unsigned*    __restrict__ cb_bits,
                                unsigned*    __restrict__ bpw) {
    int gid = blockIdx.x * 256 + threadIdx.x;
    if (gid < BINS * NWP) {
        int r = gid / NWP, w = gid % NWP;
        unsigned word = 0;
        #pragma unroll 4
        for (int b = 0; b < 32; ++b) {
            int d = w * 32 + b;
            if (d < DN && cbf[(size_t)r * DN + d] > 0.5f) word |= (1u << b);
        }
        cb_bits[gid] = word;
    } else {
        int idx = gid - BINS * NWP;
        if (idx < NCHK * 256) {
            int chunk = idx >> 8;
            int rem   = idx & 255;
            int lane  = rem >> 3;
            int j     = rem & 7;
            int half  = lane >> 4, N = lane & 15;
            unsigned word = 0;
            #pragma unroll
            for (int t = 0; t < 4; ++t) {
                int K = ((j >> 2) << 5) + (half << 4) + ((j & 3) << 2) + t;
                int d = chunk * 64 + K;
                unsigned byte = 0u;
                if (N < CN && d < DN)
                    byte = (chf[(size_t)N * DN + d] > 0.5f) ? 0x01u : 0xFFu;
                word |= byte << (8 * t);
            }
            bpw[idx] = word;
        }
    }
}

__global__ void hdc_class_pop(const float* __restrict__ chf,
                              int* __restrict__ pc) {
    __shared__ int sp[256];
    int c = blockIdx.x, tid = threadIdx.x;
    int cnt = 0;
    for (int d = tid; d < DN; d += 256)
        cnt += (chf[(size_t)c * DN + d] > 0.5f) ? 1 : 0;
    sp[tid] = cnt;
    __syncthreads();
    for (int s = 128; s > 0; s >>= 1) {
        if (tid < s) sp[tid] += sp[tid + s];
        __syncthreads();
    }
    if (tid == 0) pc[c] = sp[0];
}

__global__ void hdc_minmax_part(const float* __restrict__ x, int n,
                                float* __restrict__ pmin,
                                float* __restrict__ pmax) {
    __shared__ float smn[256], smx[256];
    int tid = threadIdx.x;
    float mn = 3.4028235e38f, mx = -3.4028235e38f;
    for (int i = blockIdx.x * 256 + tid; i < n; i += 256 * gridDim.x) {
        float t = rintf(x[i] * 100.0f) / 100.0f;
        mn = fminf(mn, t);
        mx = fmaxf(mx, t);
    }
    smn[tid] = mn; smx[tid] = mx;
    __syncthreads();
    for (int s = 128; s > 0; s >>= 1) {
        if (tid < s) {
            smn[tid] = fminf(smn[tid], smn[tid + s]);
            smx[tid] = fmaxf(smx[tid], smx[tid + s]);
        }
        __syncthreads();
    }
    if (tid == 0) { pmin[blockIdx.x] = smn[0]; pmax[blockIdx.x] = smx[0]; }
}

__global__ void hdc_finalize_edges(const float* __restrict__ pmin,
                                   const float* __restrict__ pmax,
                                   float* __restrict__ edges) {
    __shared__ float smn[256], smx[256];
    int tid = threadIdx.x;
    smn[tid] = pmin[tid];
    smx[tid] = pmax[tid];
    __syncthreads();
    for (int s = 128; s > 0; s >>= 1) {
        if (tid < s) {
            smn[tid] = fminf(smn[tid], smn[tid + s]);
            smx[tid] = fmaxf(smx[tid], smx[tid + s]);
        }
        __syncthreads();
    }
    if (tid < 16) {
        float mn = smn[0], mx = smx[0];
        float step = (mx - mn) / 15.0f;
        edges[tid] = mn + step * (float)tid;
    }
}

__global__ __launch_bounds__(64)
void hdc_bind_sim_kernel(const float*    __restrict__ x,
                         const unsigned* __restrict__ cb_bits,
                         const uint8_t*  __restrict__ bp,
                         const float*    __restrict__ edges,
                         const int*      __restrict__ pc,
                         float*          __restrict__ out) {
    __shared__ unsigned s_cb[BINS * NWP];
    __shared__ unsigned s_bnd[2][16][BND_STRIDE];
    __shared__ float    s_edges[16];

    const int tid  = threadIdx.x;
    const int wv   = tid >> 5;
    const int lane = tid & 31;
    const int tile = blockIdx.x * 2 + wv;
    const int b0   = tile * 16;

    for (int i = tid; i < BINS * NWP; i += 64) s_cb[i] = cb_bits[i];
    if (tid < 16) s_edges[tid] = edges[tid];
    __syncthreads();

    unsigned m[16];
    #pragma unroll
    for (int s = 0; s < 16; ++s) {
        const float* row = x + (size_t)(b0 + s) * FN;
        unsigned pm = 0;
        for (int f = lane; f < FN; f += 32) {
            float t = rintf(row[f] * 100.0f) / 100.0f;
            int q = -1;
            #pragma unroll
            for (int e = 0; e < 16; ++e) q += (s_edges[e] < t) ? 1 : 0;
            if (q < 0) q = 0;
            pm ^= (1u << q);
        }
        #pragma unroll
        for (int off = 16; off > 0; off >>= 1) pm ^= __shfl_xor(pm, off, 32);
        m[s] = pm;
    }

    for (int w = lane; w < NWP; w += 32) {
        unsigned c[16];
        #pragma unroll
        for (int bin = 0; bin < 16; ++bin) c[bin] = s_cb[bin * NWP + w];
        #pragma unroll
        for (int s = 0; s < 16; ++s) {
            unsigned bw = 0;
            #pragma unroll
            for (int bin = 0; bin < 16; ++bin)
                bw ^= c[bin] & (0u - ((m[s] >> bin) & 1u));
            s_bnd[wv][s][w] = bw;
        }
    }
    __syncthreads();

    const int half = lane >> 4;
    const int M    = lane & 15;
    v8i acc = {};
    #pragma unroll 2
    for (int chunk = 0; chunk < NCHK; ++chunk) {
        unsigned u0 = s_bnd[wv][M][2 * chunk];
        unsigned u1 = s_bnd[wv][M][2 * chunk + 1];
        v8i a8;
        #pragma unroll
        for (int j = 0; j < 8; ++j) {
            unsigned u    = (j < 4) ? u0 : u1;
            int      base = (half << 4) + ((j & 3) << 2);
            unsigned nib  = (u >> base) & 0xFu;
            a8[j] = (int)((nib * 0x00204081u) & 0x01010101u);
        }
        const v8i* bptr = (const v8i*)(bp + (size_t)chunk * 1024 + lane * 32);
        v8i b8 = *bptr;
        acc = __builtin_amdgcn_wmma_i32_16x16x64_iu8(
             false, a8,  true, b8, acc,
             false,  false);
    }

    const int N = lane & 15;
    if (N < CN) {
        const int base = DN - pc[N];
        #pragma unroll
        for (int r = 0; r < 8; ++r) {
            int Mrow = r + half * 8;
            out[(size_t)(b0 + Mrow) * CN + N] = (float)(base + acc[r]);
        }
    }
}

extern "C" void kernel_launch(void* const* d_in, const int* in_sizes, int n_in,
                              void* d_out, int out_size, void* d_ws, size_t ws_size,
                              hipStream_t stream) {
    const float* x   = (const float*)d_in[0];
    const float* cbf = (const float*)d_in[1];
    const float* chf = (const float*)d_in[2];
    float* out = (float*)d_out;

    uint8_t* ws = (uint8_t*)d_ws;
    unsigned* cb_bits = (unsigned*)ws;
    uint8_t*  bp      = ws + (size_t)BINS * NWP * 4;
    float*    pmin    = (float*)(ws + 20480 + 160768);
    float*    pmax    = pmin + 256;
    float*    edges   = pmax + 256;
    int*      pc      = (int*)(edges + 16);

    const int n_x = in_sizes[0];

    int pack_words = BINS * NWP + NCHK * 256;
    hdc_pack_kernel<<<(pack_words + 255) / 256, 256, 0, stream>>>(
        cbf, chf, cb_bits, (unsigned*)bp);

    hdc_class_pop<<<CN, 256, 0, stream>>>(chf, pc);
    hdc_minmax_part<<<256, 256, 0, stream>>>(x, n_x, pmin, pmax);
    hdc_finalize_edges<<<1, 256, 0, stream>>>(pmin, pmax, edges);

    hdc_bind_sim_kernel<<<128, 64, 0, stream>>>(x, cb_bits, bp, edges, pc, out);
}
